// SelfAttention_3728031613057
// MI455X (gfx1250) — compile-verified, matrix operand order checked by reading, not run on hardware
//
#include <hip/hip_runtime.h>
#include <hip/hip_bf16.h>

typedef __attribute__((ext_vector_type(16))) __bf16 v16bf;
typedef __attribute__((ext_vector_type(8)))  float  v8f;
typedef unsigned short ushort_t;

#define B_  4
#define C_  256
#define N_  4096
#define DK_ 32
#define JC_ 64   // key-chunk size in the flash loop

union BF16Tile { uint4 u[2]; v16bf v; };

__device__ inline ushort_t f2bf(float f) {
  union { float f; unsigned u; } x; x.f = f;
  unsigned r = x.u + 0x7FFFu + ((x.u >> 16) & 1u);
  return (ushort_t)(r >> 16);
}

// Load a 16x32 bf16 tile (A-layout; B-layout is the lane-symmetric mirror,
// same pattern) from row-major storage with row stride `row_stride` elements.
// Row of 32 bf16 = 16 dwords. lane<16 takes dword-pairs {0..3, 8..11} of row
// (lane&15); lane>=16 takes {4..7, 12..15}.
__device__ inline void load_tile(const ushort_t* base, int row_stride, int lane, BF16Tile& t) {
  const ushort_t* row = base + (size_t)(lane & 15) * (size_t)row_stride;
  const int half = (lane >> 4) & 1;
  const uint4* p = (const uint4*)row;
  t.u[0] = p[half];
  t.u[1] = p[2 + half];
}

#define WMMA_BF16(A, Bm, Cm) \
  __builtin_amdgcn_wmma_f32_16x16x32_bf16(false, (A), false, (Bm), (short)0, (Cm), false, false)

// --------------------------- async LDS copy path ---------------------------
// Use inline asm (avoids clang builtin pointer address-space Sema issues).
#if defined(__HIP_DEVICE_COMPILE__) && defined(__gfx1250__)
#define USE_ASYNC_LDS 1
#else
#define USE_ASYNC_LDS 0
#endif

__device__ inline void async_copy16(const ushort_t* g, ushort_t* l) {
#if USE_ASYNC_LDS
  unsigned lds_addr = (unsigned)(size_t)l;               // low 32 bits = LDS offset
  unsigned long long gaddr = (unsigned long long)(size_t)g;
  asm volatile("global_load_async_to_lds_b128 %0, %1, off"
               :
               : "v"(lds_addr), "v"(gaddr)
               : "memory");
#else
  *(uint4*)l = *(const uint4*)g;
#endif
}

__device__ inline void wait_async() {
#if USE_ASYNC_LDS
  asm volatile("s_wait_asynccnt 0x0" ::: "memory");
#endif
}

// ---------------------------------------------------------------------------
// Pre-pass 1: weights fp32 -> bf16
// ---------------------------------------------------------------------------
__global__ void convert_weights_kernel(const float* __restrict__ wq,
                                       const float* __restrict__ wk,
                                       const float* __restrict__ wv,
                                       ushort_t* __restrict__ wqb,
                                       ushort_t* __restrict__ wkb,
                                       ushort_t* __restrict__ wvb) {
  int i = blockIdx.x * blockDim.x + threadIdx.x;
  if (i < DK_ * C_) { wqb[i] = f2bf(wq[i]); wkb[i] = f2bf(wk[i]); }
  if (i < C_ * C_)  { wvb[i] = f2bf(wv[i]); }
}

// ---------------------------------------------------------------------------
// Pre-pass 2: xt[b][n][c] = bf16(x[b][c][n])  (LDS-tiled transpose)
// grid (N/32, C/32, B), block 256
// ---------------------------------------------------------------------------
__global__ __launch_bounds__(256) void transpose_kernel(const float* __restrict__ x,
                                                        ushort_t* __restrict__ xt) {
  __shared__ float tile[32][33];
  const int tx = threadIdx.x & 31;
  const int ty = threadIdx.x >> 5;   // 0..7
  const int n0 = blockIdx.x * 32;
  const int c0 = blockIdx.y * 32;
  const int b  = blockIdx.z;
#pragma unroll
  for (int k = 0; k < 4; ++k) {
    int c = c0 + ty + k * 8;
    tile[ty + k * 8][tx] = x[((size_t)b * C_ + c) * (size_t)N_ + n0 + tx];
  }
  __syncthreads();
#pragma unroll
  for (int k = 0; k < 4; ++k) {
    int n = n0 + ty + k * 8;
    xt[((size_t)b * N_ + n) * (size_t)C_ + c0 + tx] = f2bf(tile[tx][ty + k * 8]);
  }
}

// ---------------------------------------------------------------------------
// Projection Q,K: q[b][n][d], k[b][n][d] (row-per-pixel, dk=32 contiguous)
// One wave per 16-pixel tile. M=n(16), N=d(16x2), K=c(256 in chunks of 32).
// grid (N/16, B), block 32
// ---------------------------------------------------------------------------
__global__ __launch_bounds__(32) void proj_qk_kernel(const ushort_t* __restrict__ xt,
                                                     const ushort_t* __restrict__ wqb,
                                                     const ushort_t* __restrict__ wkb,
                                                     const float* __restrict__ bq,
                                                     const float* __restrict__ bk,
                                                     ushort_t* __restrict__ qo,
                                                     ushort_t* __restrict__ ko) {
  const int lane = threadIdx.x & 31;
  const int half = (lane >> 4) & 1;
  const int col  = lane & 15;
  const int b  = blockIdx.y;
  const int n0 = blockIdx.x * 16;

  v8f q0 = {}, q1 = {}, k0 = {}, k1 = {};
  for (int c0 = 0; c0 < C_; c0 += 32) {
    BF16Tile xa;
    load_tile(xt + ((size_t)b * N_ + n0) * (size_t)C_ + c0, C_, lane, xa);
    BF16Tile w;
    load_tile(wqb + c0,            C_, lane, w); q0 = WMMA_BF16(xa.v, w.v, q0);
    load_tile(wqb + 16 * C_ + c0,  C_, lane, w); q1 = WMMA_BF16(xa.v, w.v, q1);
    load_tile(wkb + c0,            C_, lane, w); k0 = WMMA_BF16(xa.v, w.v, k0);
    load_tile(wkb + 16 * C_ + c0,  C_, lane, w); k1 = WMMA_BF16(xa.v, w.v, k1);
  }
  const float bq0 = bq[col], bq1 = bq[16 + col];
  const float bk0 = bk[col], bk1 = bk[16 + col];
#pragma unroll
  for (int v = 0; v < 8; ++v) {
    int n = n0 + v + 8 * half;
    ushort_t* qrow = qo + ((size_t)b * N_ + n) * DK_;
    ushort_t* krow = ko + ((size_t)b * N_ + n) * DK_;
    qrow[col]      = f2bf(q0[v] + bq0);
    qrow[16 + col] = f2bf(q1[v] + bq1);
    krow[col]      = f2bf(k0[v] + bk0);
    krow[16 + col] = f2bf(k1[v] + bk1);
  }
}

// ---------------------------------------------------------------------------
// Projection V: v[b][e][n] bf16.  M=e(16), N=n(4x16), K=c(256).
// grid (N/64, C/16, B), block 32
// ---------------------------------------------------------------------------
__global__ __launch_bounds__(32) void proj_v_kernel(const ushort_t* __restrict__ xt,
                                                    const ushort_t* __restrict__ wvb,
                                                    const float* __restrict__ bv,
                                                    ushort_t* __restrict__ vo) {
  const int lane = threadIdx.x & 31;
  const int half = (lane >> 4) & 1;
  const int col  = lane & 15;
  const int n0 = blockIdx.x * 64;
  const int e0 = blockIdx.y * 16;
  const int b  = blockIdx.z;

  v8f acc[4];
#pragma unroll
  for (int t = 0; t < 4; ++t) acc[t] = {};

  for (int c0 = 0; c0 < C_; c0 += 32) {
    BF16Tile wa;
    load_tile(wvb + (size_t)e0 * C_ + c0, C_, lane, wa);
#pragma unroll
    for (int t = 0; t < 4; ++t) {
      BF16Tile xb;
      load_tile(xt + ((size_t)b * N_ + n0 + t * 16) * (size_t)C_ + c0, C_, lane, xb);
      acc[t] = WMMA_BF16(wa.v, xb.v, acc[t]);
    }
  }
#pragma unroll
  for (int v = 0; v < 8; ++v) {
    int e = e0 + v + 8 * half;
    float be = bv[e];
#pragma unroll
    for (int t = 0; t < 4; ++t) {
      int n = n0 + t * 16 + col;
      vo[((size_t)b * C_ + e) * (size_t)N_ + n] = f2bf(acc[t][v] + be);
    }
  }
}

// ---------------------------------------------------------------------------
// Fused flash attention + residual, double-buffered async-LDS staging.
// Block = 4 waves; wave w owns query tile n0 = blockIdx.x*64 + w*16.
// Key chunks of 64: per chunk per wave 4 energy WMMAs + 32 PV WMMAs.
// grid (N/64, B), block 128
// ---------------------------------------------------------------------------
__device__ inline void stage_chunk(const ushort_t* kbb, const ushort_t* vbb, int j0,
                                   int tid, ushort_t* sKbuf, ushort_t* sVbuf) {
  // K chunk: 64 rows x 32 bf16 (64B/row) = 256 x b128 -> 2 per thread
#pragma unroll
  for (int r = 0; r < 2; ++r) {
    int idx = tid + r * 128;
    int row = idx >> 2, seg = idx & 3;
    async_copy16(kbb + (size_t)(j0 + row) * DK_ + seg * 8, sKbuf + row * DK_ + seg * 8);
  }
  // V chunk: 256 rows x 64 bf16 (128B/row) = 2048 x b128 -> 16 per thread
#pragma unroll
  for (int r = 0; r < 16; ++r) {
    int idx = tid + r * 128;
    int row = idx >> 3, seg = idx & 7;
    async_copy16(vbb + (size_t)row * N_ + j0 + seg * 8, sVbuf + (size_t)row * JC_ + seg * 8);
  }
}

__global__ __launch_bounds__(128) void attn_kernel(const ushort_t* __restrict__ qb,
                                                   const ushort_t* __restrict__ kb,
                                                   const ushort_t* __restrict__ vb,
                                                   const float* __restrict__ x,
                                                   const float* __restrict__ gamma,
                                                   float* __restrict__ out) {
  __shared__ ushort_t sK[2][JC_ * DK_];   // 2 x 64 keys x 32 dk      (2 x 4 KB)
  __shared__ ushort_t sV[2][C_ * JC_];    // 2 x 256 ch  x 64 keys    (2 x 32 KB)
  __shared__ ushort_t sP[4][16 * JC_];    // per-wave P tile 16 x 64  (4 x 2 KB)

  const int tid  = threadIdx.x;
  const int lane = tid & 31;
  const int wave = tid >> 5;
  const int half = (lane >> 4) & 1;
  const int col  = lane & 15;
  const int b  = blockIdx.y;
  const int n0 = blockIdx.x * 64 + wave * 16;

  const ushort_t* kbb = kb + (size_t)b * N_ * DK_;
  const ushort_t* vbb = vb + (size_t)b * C_ * N_;

  BF16Tile qA;
  load_tile(qb + ((size_t)b * N_ + n0) * DK_, DK_, lane, qA);

  float m[8], l[8];
  v8f acc[16];
#pragma unroll
  for (int v = 0; v < 8; ++v) { m[v] = -1e30f; l[v] = 0.0f; }
#pragma unroll
  for (int t = 0; t < 16; ++t) acc[t] = {};

  // prologue: stage chunk 0 into buffer 0
  stage_chunk(kbb, vbb, 0, tid, sK[0], sV[0]);

  const int NCHUNK = N_ / JC_;   // 64
  for (int i = 0; i < NCHUNK; ++i) {
    const int cur = i & 1;
    const int nxt = cur ^ 1;

    wait_async();        // this wave's copies into sK/sV[cur] are complete
    __syncthreads();     // all waves' copies complete; prior reads of [nxt] done

    if (i + 1 < NCHUNK)
      stage_chunk(kbb, vbb, (i + 1) * JC_, tid, sK[nxt], sV[nxt]);

    const ushort_t* sKc = sK[cur];
    const ushort_t* sVc = sV[cur];

    // energy: 16 queries x 64 keys, K=dk=32 -> four WMMAs
    v8f e[4];
    v8f z = {};
#pragma unroll
    for (int u = 0; u < 4; ++u) {
      BF16Tile kB;
      load_tile(sKc + u * 16 * DK_, DK_, lane, kB);
      e[u] = WMMA_BF16(qA.v, kB.v, z);
    }

    // online softmax; C-layout: VGPR v, lane -> row v + 8*half, col = lane&15
    float sc[8];
    ushort_t* sp = sP[wave];
#pragma unroll
    for (int v = 0; v < 8; ++v) {
      float a0 = e[0][v], a1 = e[1][v], a2 = e[2][v], a3 = e[3][v];
      float mx = fmaxf(fmaxf(a0, a1), fmaxf(a2, a3));
      mx = fmaxf(mx, __shfl_xor(mx, 1));
      mx = fmaxf(mx, __shfl_xor(mx, 2));
      mx = fmaxf(mx, __shfl_xor(mx, 4));
      mx = fmaxf(mx, __shfl_xor(mx, 8));
      float mn = fmaxf(m[v], mx);
      float s  = __expf(m[v] - mn);
      float p0 = __expf(a0 - mn);
      float p1 = __expf(a1 - mn);
      float p2 = __expf(a2 - mn);
      float p3 = __expf(a3 - mn);
      float rs = (p0 + p1) + (p2 + p3);
      rs += __shfl_xor(rs, 1);
      rs += __shfl_xor(rs, 2);
      rs += __shfl_xor(rs, 4);
      rs += __shfl_xor(rs, 8);
      m[v] = mn;
      l[v] = l[v] * s + rs;
      sc[v] = s;
      ushort_t* r = sp + (v + 8 * half) * JC_;
      r[col]      = f2bf(p0);
      r[16 + col] = f2bf(p1);
      r[32 + col] = f2bf(p2);
      r[48 + col] = f2bf(p3);
    }
    // rescale accumulators (row mapping matches energy tiles)
#pragma unroll
    for (int t = 0; t < 16; ++t)
#pragma unroll
      for (int v = 0; v < 8; ++v) acc[t][v] *= sc[v];

    // P * V^T : A = P (16q x 64j, two K=32 halves), B = V rows per channel tile
    BF16Tile pA0, pA1;
    load_tile(sp,      JC_, lane, pA0);
    load_tile(sp + 32, JC_, lane, pA1);
#pragma unroll
    for (int t = 0; t < 16; ++t) {
      BF16Tile vB0, vB1;
      load_tile(sVc + (size_t)(t * 16) * JC_,      JC_, lane, vB0);
      load_tile(sVc + (size_t)(t * 16) * JC_ + 32, JC_, lane, vB1);
      acc[t] = WMMA_BF16(pA0.v, vB0.v, acc[t]);
      acc[t] = WMMA_BF16(pA1.v, vB1.v, acc[t]);
    }
  }

  const float g = gamma[0];
  float inv[8];
#pragma unroll
  for (int v = 0; v < 8; ++v) inv[v] = 1.0f / l[v];
#pragma unroll
  for (int t = 0; t < 16; ++t) {
#pragma unroll
    for (int v = 0; v < 8; ++v) {
      int n = n0 + v + 8 * half;
      int c = t * 16 + col;
      size_t o = ((size_t)b * C_ + c) * (size_t)N_ + n;
      out[o] = g * (acc[t][v] * inv[v]) + x[o];
    }
  }
}

// ---------------------------------------------------------------------------
extern "C" void kernel_launch(void* const* d_in, const int* in_sizes, int n_in,
                              void* d_out, int out_size, void* d_ws, size_t ws_size,
                              hipStream_t stream) {
  const float* x     = (const float*)d_in[0];
  const float* wq    = (const float*)d_in[1];
  const float* bq    = (const float*)d_in[2];
  const float* wk    = (const float*)d_in[3];
  const float* bk    = (const float*)d_in[4];
  const float* wv    = (const float*)d_in[5];
  const float* bv    = (const float*)d_in[6];
  const float* gamma = (const float*)d_in[7];
  float* out = (float*)d_out;

  char* ws = (char*)d_ws;
  size_t off = 0;
  ushort_t* xt  = (ushort_t*)(ws + off); off += (size_t)B_ * N_ * C_ * 2;   // 8 MB
  ushort_t* qbf = (ushort_t*)(ws + off); off += (size_t)B_ * N_ * DK_ * 2;  // 1 MB
  ushort_t* kbf = (ushort_t*)(ws + off); off += (size_t)B_ * N_ * DK_ * 2;  // 1 MB
  ushort_t* vbf = (ushort_t*)(ws + off); off += (size_t)B_ * C_ * N_ * 2;   // 8 MB
  ushort_t* wqb = (ushort_t*)(ws + off); off += (size_t)DK_ * C_ * 2;
  ushort_t* wkb = (ushort_t*)(ws + off); off += (size_t)DK_ * C_ * 2;
  ushort_t* wvb = (ushort_t*)(ws + off); off += (size_t)C_ * C_ * 2;

  convert_weights_kernel<<<(C_ * C_ + 255) / 256, 256, 0, stream>>>(wq, wk, wv, wqb, wkb, wvb);
  transpose_kernel<<<dim3(N_ / 32, C_ / 32, B_), 256, 0, stream>>>(x, xt);
  proj_qk_kernel<<<dim3(N_ / 16, B_), 32, 0, stream>>>(xt, wqb, wkb, bq, bk, qbf, kbf);
  proj_v_kernel<<<dim3(N_ / 64, C_ / 16, B_), 32, 0, stream>>>(xt, wvb, bv, vbf);
  attn_kernel<<<dim3(N_ / 64, B_), 128, 0, stream>>>(qbf, kbf, vbf, x, gamma, out);
}
